// DeformConvV2_10127532884579
// MI455X (gfx1250) — matrix operand order repaired and checked by reading, not compiled, not run on hardware
//
#include <hip/hip_runtime.h>
#include <hip/hip_bf16.h>
#include <stdint.h>

// ---------------------------------------------------------------------------
// DeformConvV2 forward for MI455X (gfx1250), bf16 WMMA pipeline:
//   k_prep    : cvt weight->bf16 [128][1152]; build conv weights cw [32][1152]
//   k_offconv : offset/mask conv as implicit-GEMM WMMA (M=32,K=1152), sigmoid
//   k_im2col  : deformable bilinear gather -> col bf16 [B][hw][1152] (75.5MB,
//               L2-resident on the 192MB L2)
//   k_gemm    : out = W x col, M=128,N=4096/batch,K=1152, TDM triple-buffered
// ---------------------------------------------------------------------------

typedef __bf16 bf16_t;
typedef __attribute__((ext_vector_type(16))) __bf16 bf16x16;
typedef __attribute__((ext_vector_type(8)))  __bf16 bf16x8;
typedef __attribute__((ext_vector_type(2)))  __bf16 bf16x2;
typedef __attribute__((ext_vector_type(8)))  float  f32x8;
typedef __attribute__((ext_vector_type(4)))  unsigned int v4u;
typedef __attribute__((ext_vector_type(8)))  int v8i;
typedef __attribute__((ext_vector_type(4)))  int v4i;

#define B_    8
#define C_    128
#define HW_   4096
#define K_    1152
#define COUT_ 128

// TDM availability: evaluate ONLY on the device pass; the host pass (x86)
// never has amdgcn builtins and must just compile a stub.
#if defined(__HIP_DEVICE_COMPILE__)
  #if __has_builtin(__builtin_amdgcn_tensor_load_to_lds)
    #define HAVE_TDM 1
  #else
    #error "gfx1250 DEVICE pass is missing __builtin_amdgcn_tensor_load_to_lds"
  #endif
#endif

__device__ __forceinline__ bf16_t f2bf(float f) { return (bf16_t)f; }

__device__ __forceinline__ f32x8 wmma_bf16(bf16x16 a, bf16x16 b, f32x8 c) {
  return __builtin_amdgcn_wmma_f32_16x16x32_bf16(false, a, false, b,
                                                 (short)0, c, false, false);
}

// A fragment: 16x32 tile of row-major [*, ld] bf16 matrix.
// ISA layout: lanes 0-15 & 16-31 both hold M=lane&15; lane-half selects which
// 8-element K runs: halves {k0+h*8, k0+16+h*8} -> two contiguous b128 loads.
__device__ __forceinline__ bf16x16 load_a_frag(const bf16_t* mat, int ld,
                                               int m0, int k0, int lane) {
  const int half = lane >> 4;
  const bf16_t* base = mat + (size_t)(m0 + (lane & 15)) * ld + k0 + half * 8;
  union { bf16x16 v; bf16x8 h[2]; } u;
  u.h[0] = *reinterpret_cast<const bf16x8*>(base);
  u.h[1] = *reinterpret_cast<const bf16x8*>(base + 16);
  return u.v;
}

// B fragment: 32x16 (KxN) tile, LDS stored N-major: Bs[n][32] so each lane
// reads bf16 runs at Bs[n0+lane&15][half*8] and [16+half*8] (two ds_load_b128).
// ISA layout: lane-half h holds K {h*8..h*8+7, 16+h*8..16+h*8+7}, N=lane&15.
__device__ __forceinline__ bf16x16 load_b_frag(const bf16_t* Bs, int n0, int lane) {
  const int half = lane >> 4;
  const bf16_t* base = Bs + (size_t)(n0 + (lane & 15)) * 32 + half * 8;
  union { bf16x16 v; bf16x8 h[2]; } u;
  u.h[0] = *reinterpret_cast<const bf16x8*>(base);
  u.h[1] = *reinterpret_cast<const bf16x8*>(base + 16);
  return u.v;
}

// ---------------------------------------------------------------------------
// Tensor Data Mover: 2D tile load, descriptor per CDNA5 ISA
// ---------------------------------------------------------------------------
__device__ __forceinline__ void tdm_load_2d(unsigned lds_addr, const void* gptr,
                                            unsigned tile_d0, unsigned tile_d1,
                                            unsigned tensor_d0, unsigned tensor_d1,
                                            unsigned stride_d0, unsigned dsz_code) {
#ifdef HAVE_TDM
  unsigned long long ga = (unsigned long long)(size_t)gptr;
  v4u g0;
  g0.x = 1u;                                            // count=1, user mode
  g0.y = lds_addr;                                      // LDS byte offset
  g0.z = (unsigned)(ga & 0xffffffffu);                  // global_addr[31:0]
  g0.w = (unsigned)((ga >> 32) & 0x1ffffffu) | (2u << 30); // addr[56:32]|type=2
  v8i g1;
  g1[0] = (int)(dsz_code << 16);                        // data_size (1 => 2B)
  g1[1] = (int)((tensor_d0 & 0xffffu) << 16);           // tensor_dim0[15:0]
  g1[2] = (int)(((tensor_d0 >> 16) & 0xffffu) | ((tensor_d1 & 0xffffu) << 16));
  g1[3] = (int)(((tensor_d1 >> 16) & 0xffffu) | (tile_d0 << 16));
  g1[4] = (int)(tile_d1 & 0xffffu);                     // tile_dim1, tile_dim2=0
  g1[5] = (int)stride_d0;                               // tensor_dim0_stride lo
  g1[6] = 0;                                            // stride hi | dim1_stride lo
  g1[7] = 0;
  v4i z4 = {0, 0, 0, 0};
#if defined(__clang_major__) && __clang_major__ >= 23
  v8i z8 = {0, 0, 0, 0, 0, 0, 0, 0};
  __builtin_amdgcn_tensor_load_to_lds(g0, g1, z4, z4, z8, 0);
#else
  __builtin_amdgcn_tensor_load_to_lds(g0, g1, z4, z4, 0);
#endif
#else
  (void)lds_addr; (void)gptr; (void)tile_d0; (void)tile_d1;
  (void)tensor_d0; (void)tensor_d1; (void)stride_d0; (void)dsz_code;
#endif
}

template <int N>
__device__ __forceinline__ void wait_tensorcnt() {
#ifdef HAVE_TDM
  __builtin_amdgcn_s_wait_tensorcnt(N);
#endif
}

// ---------------------------------------------------------------------------
// Kernel 0: weight conversion / conv-weight matrix build
// ---------------------------------------------------------------------------
__global__ __launch_bounds__(256) void k_prep(const float* __restrict__ weight,
                                              const float* __restrict__ offw,
                                              const float* __restrict__ maskw,
                                              bf16_t* __restrict__ wgt,
                                              bf16_t* __restrict__ cw) {
  int i = blockIdx.x * 256 + threadIdx.x;
  if (i < COUT_ * K_) {
    wgt[i] = f2bf(weight[i]);
  } else {
    int j = i - COUT_ * K_;
    if (j < 32 * K_) {
      int r = j / K_, q = j - r * K_;
      float v = 0.f;
      if (r < 18)      v = offw[r * K_ + q];          // (18,C,3,3) flat
      else if (r < 27) v = maskw[(r - 18) * K_ + q];  // (9,C,3,3) flat
      cw[j] = f2bf(v);
    }
  }
}

// ---------------------------------------------------------------------------
// Kernel 1: offset/mask conv, implicit-GEMM WMMA.  O[32][128px] per block.
// 8 waves: m-tile = w&1, n-block = (w>>1)*32 (two 16-wide n-tiles).
// Fill: thread owns kl=tid&31 (fixed p per chunk) and sweeps 16 pixels of one
// image row.  Two-phase: batch all 16 gathers into registers (loads stay in
// flight together), then mask/convert/store.
// ---------------------------------------------------------------------------
__global__ __launch_bounds__(256) void k_offconv(const float* __restrict__ data,
                                                 const bf16_t* __restrict__ cw,
                                                 float* __restrict__ offb,
                                                 float* __restrict__ mb) {
  __shared__ alignas(16) bf16_t s_bt[2][128 * 32];  // [buf][n][kl], N-major
  const int tid = threadIdx.x, lane = tid & 31, w = tid >> 5;
  const int b = blockIdx.x >> 5, hw0 = (blockIdx.x & 31) << 7;
  const float* datab = data + ((size_t)b * C_ << 12);

  const int kl = tid & 31;            // K-slot within chunk (fixed per thread)
  const int nb = tid >> 5;            // pixel group: n = nb*16 + i
  const int hwb = hw0 + nb * 16;      // 16-pixel run stays in one image row:
  const int yy = hwb >> 6;            //   y constant per thread
  const int xb = hwb & 63;            //   x = xb + i, xb in {0,16,32,48}

  // index state for p = kc*32 + kl  (p = c*9 + ki*3 + kj); init 3 divisions.
  int cc = kl / 9;
  int rr = kl - cc * 9;
  int kiv = rr / 3, kjv = rr - (rr / 3) * 3;

  auto fill = [&](int buf) {
    const int sy = yy + kiv - 1;
    const bool vy = (unsigned)sy < 64u;
    const int syc = sy < 0 ? 0 : (sy > 63 ? 63 : sy);
    const int rowbase = (cc << 12) + (syc << 6);
    const int sxb = xb + kjv - 1;
    bf16_t* wp = &s_bt[buf][nb * 512 + kl];          // + i*32 elem per step
    float vals[16];
    #pragma unroll
    for (int i = 0; i < 16; ++i) {                   // phase A: batch loads
      int sx = sxb + i;
      int sxc = sx < 0 ? 0 : (sx > 63 ? 63 : sx);    // med3
      vals[i] = datab[rowbase + sxc];                // saddr + 32-bit voffset
    }
    #pragma unroll
    for (int i = 0; i < 16; ++i) {                   // phase B: mask+cvt+store
      int sx = sxb + i;
      bool vv = vy && ((unsigned)sx < 64u);
      float v = vv ? vals[i] : 0.f;
      wp[i * 32] = f2bf(v);                          // ds_store_b16, imm offs
    }
    cc += 3;                                         // advance p by 32 = 3*9+5
    int r = kiv * 3 + kjv + 5;
    if (r >= 9) { r -= 9; ++cc; }
    kiv = r / 3; kjv = r - kiv * 3;
  };

  const f32x8 z8 = {0.f, 0.f, 0.f, 0.f, 0.f, 0.f, 0.f, 0.f};
  f32x8 acc[2] = {z8, z8};

  fill(0);                                     // prologue: chunk 0 -> buf 0
  for (int kc = 0; kc < 36; ++kc) {
    const int buf = kc & 1;
    const int k0 = kc * 32;
    __syncthreads();                           // buf ready; buf^1 free
    bf16x16 a   = load_a_frag(cw, K_, (w & 1) * 16, k0, lane);
    bf16x16 bf0 = load_b_frag(&s_bt[buf][0], ((w >> 1) * 2 + 0) * 16, lane);
    bf16x16 bf1 = load_b_frag(&s_bt[buf][0], ((w >> 1) * 2 + 1) * 16, lane);
    if (kc + 1 < 36) fill(buf ^ 1);            // overlap gather with WMMA
    acc[0] = wmma_bf16(a, bf0, acc[0]);
    acc[1] = wmma_bf16(a, bf1, acc[1]);
  }

  const int half = lane >> 4;
  #pragma unroll
  for (int nt = 0; nt < 2; ++nt)
    #pragma unroll
    for (int r = 0; r < 8; ++r) {
      int ch = (w & 1) * 16 + half * 8 + r;
      int ohw = hw0 + ((w >> 1) * 2 + nt) * 16 + (lane & 15);
      float v = acc[nt][r];
      if (ch < 18)
        offb[((size_t)b * 18 + ch) * HW_ + ohw] = v;
      else if (ch < 27)
        mb[((size_t)b * 9 + (ch - 18)) * HW_ + ohw] = 1.f / (1.f + __expf(-v));
    }
}

// ---------------------------------------------------------------------------
// Kernel 2: deformable im2col -> col bf16 [b][hw][p], p = c*9+k
// Block = (b, 32 pixels). Phase 1: 288 (px,k) bilinear params to LDS.
// Phase 2: thread handles pairs p = 2*tid + j*512 + {0,1}: 8 gathers batched
// per pair, one packed 4-byte bf16x2 store (p even => aligned).
// ---------------------------------------------------------------------------
__global__ __launch_bounds__(256) void k_im2col(const float* __restrict__ data,
                                                const float* __restrict__ offb,
                                                const float* __restrict__ mb,
                                                bf16_t* __restrict__ col) {
  const int tid = threadIdx.x;
  const int b = blockIdx.x >> 7, hw0 = (blockIdx.x & 127) << 5;
  const float* datab = data + ((size_t)b * C_ << 12);
  __shared__ int   s_y0[288], s_x0[288];
  __shared__ float s_ly[288], s_lx[288], s_m[288];
  if (tid < 288) {
    int px = tid / 9, k = tid - px * 9;
    int hw = hw0 + px;
    int yo = hw >> 6, xo = hw & 63;
    int ki = k / 3, kj = k - ki * 3;
    float oy = offb[((size_t)b * 18 + 2 * k) * HW_ + hw];
    float ox = offb[((size_t)b * 18 + 2 * k + 1) * HW_ + hw];
    float y = (float)(yo - 1 + ki) + oy;
    float x = (float)(xo - 1 + kj) + ox;
    float fy = floorf(y), fx = floorf(x);
    s_y0[tid] = (int)fy;  s_x0[tid] = (int)fx;
    s_ly[tid] = y - fy;   s_lx[tid] = x - fx;
    s_m[tid]  = mb[((size_t)b * 9 + k) * HW_ + hw];
  }
  __syncthreads();

  // masked bilinear sample for (c, param slot idx): 4 batched gathers
  auto sample = [&](int cbase, int idx) -> float {
    int y0 = s_y0[idx], x0 = s_x0[idx];
    int y1 = y0 + 1, x1 = x0 + 1;
    float ly = s_ly[idx], lx = s_lx[idx];
    bool vy0 = (unsigned)y0 < 64u, vy1 = (unsigned)y1 < 64u;
    bool vx0 = (unsigned)x0 < 64u, vx1 = (unsigned)x1 < 64u;
    int cy0 = y0 < 0 ? 0 : (y0 > 63 ? 63 : y0);
    int cy1 = y1 < 0 ? 0 : (y1 > 63 ? 63 : y1);
    int cx0 = x0 < 0 ? 0 : (x0 > 63 ? 63 : x0);
    int cx1 = x1 < 0 ? 0 : (x1 > 63 ? 63 : x1);
    float v00 = datab[cbase + (cy0 << 6) + cx0];
    float v01 = datab[cbase + (cy0 << 6) + cx1];
    float v10 = datab[cbase + (cy1 << 6) + cx0];
    float v11 = datab[cbase + (cy1 << 6) + cx1];
    v00 = (vy0 && vx0) ? v00 : 0.f;
    v01 = (vy0 && vx1) ? v01 : 0.f;
    v10 = (vy1 && vx0) ? v10 : 0.f;
    v11 = (vy1 && vx1) ? v11 : 0.f;
    float v = (1.f - ly) * (1.f - lx) * v00 + (1.f - ly) * lx * v01 +
              ly * (1.f - lx) * v10 + ly * lx * v11;
    return v * s_m[idx];
  };

  // (c,k) for even p = 2*tid + j*512:  +512 => c += 56, k += 8 (mod 9)
  int c0A[3], k0A[3], c1A[3], k1A[3];
  {
    int p0 = 2 * tid;
    int cc = p0 / 9, kv = p0 - (p0 / 9) * 9;
    #pragma unroll
    for (int j = 0; j < 3; ++j) {
      c0A[j] = cc; k0A[j] = kv;
      c1A[j] = (kv == 8) ? cc + 1 : cc;          // p+1
      k1A[j] = (kv == 8) ? 0 : kv + 1;
      cc += 56; kv += 8;
      if (kv >= 9) { kv -= 9; ++cc; }
    }
  }

  for (int px = 0; px < 32; ++px) {
    bf16_t* row = col + ((size_t)b * HW_ + hw0 + px) * (size_t)K_;
    const int pxb = px * 9;
    #pragma unroll
    for (int j = 0; j < 3; ++j) {
      int p = 2 * tid + j * 512;
      if (j < 2 || p < K_) {                      // j=2 valid only for tid<64
        float v0 = sample(c0A[j] << 12, pxb + k0A[j]);
        float v1 = sample(c1A[j] << 12, pxb + k1A[j]);
        bf16x2 pr;
        pr[0] = f2bf(v0);
        pr[1] = f2bf(v1);
        *reinterpret_cast<bf16x2*>(row + p) = pr; // one b32 store per pair
      }
    }
  }
}

// ---------------------------------------------------------------------------
// Kernel 3: main GEMM. Block = (b, 128 hw-tile), 128x128 output, K=1152.
// B tile (128 hw x 32 k, K-contiguous) fetched by TDM into LDS, TRIPLE
// buffered with one barrier per iteration: wait tensorcnt<=1 (two chunks in
// flight, in-order completion => current chunk done), barrier, then issue
// chunk kc+2 into the slot everyone finished reading in iteration kc-1.
// 8 waves: m-tiles {(w&3)*2, +1}, n-tiles {(w>>2)*4 .. +3} -> 8 WMMA/iter/wave.
// ---------------------------------------------------------------------------
__global__ __launch_bounds__(256) void k_gemm(const bf16_t* __restrict__ wgt,
                                              const bf16_t* __restrict__ col,
                                              float* __restrict__ out) {
  __shared__ alignas(16) bf16_t s_b[3][128 * 32];
  const int tid = threadIdx.x, lane = tid & 31, w = tid >> 5;
  const int b = blockIdx.x >> 5, hw0 = (blockIdx.x & 31) << 7;
  const bool lead = (__builtin_amdgcn_readfirstlane((int)tid) == 0);
  const bf16_t* colb = col + ((size_t)b * HW_ + hw0) * (size_t)K_;
  const f32x8 z8 = {0.f, 0.f, 0.f, 0.f, 0.f, 0.f, 0.f, 0.f};
  f32x8 acc[2][4];
  #pragma unroll
  for (int mt = 0; mt < 2; ++mt)
    #pragma unroll
    for (int nt = 0; nt < 4; ++nt) acc[mt][nt] = z8;

  if (lead) {                                   // prologue: chunks 0,1 queued
    tdm_load_2d((unsigned)(size_t)&s_b[0][0], colb, 32u, 128u,
                (unsigned)K_, (unsigned)HW_, (unsigned)K_, 1u);
    tdm_load_2d((unsigned)(size_t)&s_b[1][0], colb + 32, 32u, 128u,
                (unsigned)K_, (unsigned)HW_, (unsigned)K_, 1u);
  }

  const int mrow0 = ((w & 3) * 2 + 0) * 16;
  const int mrow1 = ((w & 3) * 2 + 1) * 16;

  int slot = 0;                                 // kc % 3, maintained
  for (int kc = 0; kc < 36; ++kc) {
    const int k0 = kc * 32;
    if (lead) {
      if (kc + 1 < 36) wait_tensorcnt<1>();     // chunk kc done (in-order)
      else             wait_tensorcnt<0>();
    }
    __syncthreads();                            // tile kc visible; slot
                                                // (kc+2)%3 fully consumed
    if (lead && kc + 2 < 36) {
      int isl = slot == 0 ? 2 : slot - 1;       // (kc+2)%3
      tdm_load_2d((unsigned)(size_t)&s_b[isl][0], colb + (k0 + 64),
                  32u, 128u, (unsigned)K_, (unsigned)HW_, (unsigned)K_, 1u);
    }
    bf16x16 a0 = load_a_frag(wgt, K_, mrow0, k0, lane);
    bf16x16 a1 = load_a_frag(wgt, K_, mrow1, k0, lane);
    #pragma unroll
    for (int nt = 0; nt < 4; ++nt) {
      bf16x16 bf = load_b_frag(&s_b[slot][0], ((w >> 2) * 4 + nt) * 16, lane);
      acc[0][nt] = wmma_bf16(a0, bf, acc[0][nt]);
      acc[1][nt] = wmma_bf16(a1, bf, acc[1][nt]);
    }
    slot = slot == 2 ? 0 : slot + 1;
  }

  const int half = lane >> 4;
  float* outb = out + ((size_t)b * COUT_) * HW_ + hw0;
  #pragma unroll
  for (int mt = 0; mt < 2; ++mt)
    #pragma unroll
    for (int nt = 0; nt < 4; ++nt)
      #pragma unroll
      for (int r = 0; r < 8; ++r) {
        int o = ((w & 3) * 2 + mt) * 16 + half * 8 + r;
        int n = ((w >> 2) * 4 + nt) * 16 + (lane & 15);
        outb[(size_t)o * HW_ + n] = acc[mt][nt][r];
      }
}

// ---------------------------------------------------------------------------
extern "C" void kernel_launch(void* const* d_in, const int* in_sizes, int n_in,
                              void* d_out, int out_size, void* d_ws, size_t ws_size,
                              hipStream_t stream) {
  (void)in_sizes; (void)n_in; (void)out_size; (void)ws_size;
  const float* data   = (const float*)d_in[0];  // (8,128,64,64)
  const float* weight = (const float*)d_in[1];  // (128,1152)
  const float* offw   = (const float*)d_in[2];  // (18,128,3,3)
  const float* maskw  = (const float*)d_in[3];  // (9,128,3,3)
  float* out = (float*)d_out;                   // (8,128,64,64)

  char* ws = (char*)d_ws;                       // workspace layout (bytes):
  bf16_t* wgt  = (bf16_t*)(ws + 0);             //   294912  : weight bf16
  bf16_t* cw   = (bf16_t*)(ws + 294912);        //    73728  : conv wgt 32x1152
  float*  offb = (float*)(ws + 368640);         //  2359296  : offsets f32
  float*  mb   = (float*)(ws + 2727936);        //  1179648  : mask f32
  bf16_t* col  = (bf16_t*)(ws + 3907584);       // 75497472  : col bf16

  k_prep   <<<720,  256, 0, stream>>>(weight, offw, maskw, wgt, cw);
  k_offconv<<<256,  256, 0, stream>>>(data, cw, offb, mb);
  k_im2col <<<1024, 256, 0, stream>>>(data, offb, mb, col);
  k_gemm   <<<256,  256, 0, stream>>>(wgt, col, out);
}
